// MultiNetworkLinear_41695542509718
// MI455X (gfx1250) — matrix operand order repaired and checked by reading, not compiled, not run on hardware
//
#include <hip/hip_runtime.h>
#include <hip/hip_bf16.h>

#define NNET 512
#define KDIM 256
#define ODIM 256
#define NPTS 262144
#define BM   128
#define BN   128
#define KC   64
#define LDSS 72   // LDS row stride in bf16 elems: 144B = 36 dwords (16B aligned, conflict-free)

typedef __bf16 bf16;
typedef bf16  bf16x4 __attribute__((ext_vector_type(4)));
typedef bf16  bf16x8 __attribute__((ext_vector_type(8)));
typedef bf16  v16bf  __attribute__((ext_vector_type(16)));
typedef float v8f    __attribute__((ext_vector_type(8)));
typedef float v4f    __attribute__((ext_vector_type(4)));

static __device__ __forceinline__ v16bf join8(bf16x8 a, bf16x8 b) {
    return __builtin_shufflevector(a, b, 0,1,2,3,4,5,6,7,8,9,10,11,12,13,14,15);
}

// ---- pass 1: per-network segment offsets via binary search on sorted net_id ----
__global__ void seg_offsets_kernel(const int* __restrict__ net_id, int* __restrict__ offs) {
    int n = blockIdx.x * blockDim.x + threadIdx.x;
    if (n > NNET) return;
    int lo = 0, hi = NPTS;
    while (lo < hi) {
        int mid = (lo + hi) >> 1;
        if (net_id[mid] < n) lo = mid + 1; else hi = mid;
    }
    offs[n] = lo;   // offs[512] == NPTS
}

// ---- pass 2: grouped GEMM with bf16x3 split on WMMA ----
__launch_bounds__(256, 1)
__global__ void grouped_linear_kernel(const float* __restrict__ x,
                                      const float* __restrict__ weight,
                                      const float* __restrict__ bias,
                                      const int* __restrict__ offs,
                                      float* __restrict__ out) {
    __shared__ bf16 XH[BM * LDSS];
    __shared__ bf16 XL[BM * LDSS];
    __shared__ bf16 WH[BN * LDSS];
    __shared__ bf16 WL[BN * LDSS];

    const int n  = blockIdx.x;
    const int m0 = blockIdx.y * BM;
    const int o0 = blockIdx.z * BN;

    const int seg_beg = offs[n];
    const int seg_end = offs[n + 1];
    const int seg_len = seg_end - seg_beg;
    if (m0 >= seg_len) return;

    const int tid  = threadIdx.x;
    const int wave = tid >> 5;        // 0..7
    const int lane = tid & 31;
    const int wm   = wave >> 1;       // 0..3 : rows [wm*32, wm*32+32)
    const int wn   = wave & 1;        // 0..1 : cols [wn*64, wn*64+64)
    const int hs   = lane >> 4;       // half-wave select
    const int l15  = lane & 15;

    const float* wbase = weight + (size_t)n * KDIM * ODIM;

    // accumulators, bias-initialized (bias depends only on N -> same value in all 8 VGPRs)
    v8f acc[2][4];
#pragma unroll
    for (int nt = 0; nt < 4; ++nt) {
        float bv = bias[n * ODIM + o0 + wn * 64 + nt * 16 + l15];
        v8f c;
#pragma unroll
        for (int r = 0; r < 8; ++r) c[r] = bv;
        acc[0][nt] = c;
        acc[1][nt] = c;
    }

    for (int kc = 0; kc < KDIM; kc += KC) {
        __syncthreads();
        // ---- X tile [BM x KC] fp32 -> bf16 hi/lo, row-major LDS ----
#pragma unroll
        for (int i = 0; i < 8; ++i) {
            int g    = tid + i * 256;     // 0..2047 float4-groups
            int row  = g >> 4;            // 0..127
            int c4   = g & 15;            // float4 column group
            int grow = seg_beg + m0 + row;
            v4f v = {0.f, 0.f, 0.f, 0.f};
            if (grow < seg_end)
                v = *(const v4f*)(x + (size_t)grow * KDIM + kc + c4 * 4);
            bf16x4 h, l;
#pragma unroll
            for (int q = 0; q < 4; ++q) {
                bf16 hq = (bf16)v[q];
                h[q] = hq;
                l[q] = (bf16)(v[q] - (float)hq);
            }
            *(bf16x4*)&XH[row * LDSS + c4 * 4] = h;
            *(bf16x4*)&XL[row * LDSS + c4 * 4] = l;
        }
        // ---- W tile [KC x BN] fp32 -> transposed bf16 hi/lo LDS [BN x KC] ----
        // each thread: 4 k-consecutive scalars at one o (coalesced per row), one b64 store
#pragma unroll
        for (int i = 0; i < 8; ++i) {
            int idx = tid + i * 256;      // 0..2047
            int oc  = idx & 127;          // 0..127
            int kq  = idx >> 7;           // 0..15  -> k base kq*4
            bf16x4 h, l;
#pragma unroll
            for (int q = 0; q < 4; ++q) {
                float f = wbase[(size_t)(kc + kq * 4 + q) * ODIM + o0 + oc];
                bf16 hq = (bf16)f;
                h[q] = hq;
                l[q] = (bf16)(f - (float)hq);
            }
            *(bf16x4*)&WH[oc * LDSS + kq * 4] = h;
            *(bf16x4*)&WL[oc * LDSS + kq * 4] = l;
        }
        __syncthreads();

        // ---- compute: two k-steps of 32 ----
#pragma unroll
        for (int ks = 0; ks < KC; ks += 32) {
            // A frags (16x32 bf16): lanes 0-15 -> K[0..7],[16..23]; lanes 16-31 -> K[8..15],[24..31]
            v16bf ahi[2], alo[2];
#pragma unroll
            for (int mt = 0; mt < 2; ++mt) {
                int row  = wm * 32 + mt * 16 + l15;
                int base = row * LDSS + ks + hs * 8;
                ahi[mt] = join8(*(const bf16x8*)&XH[base], *(const bf16x8*)&XH[base + 16]);
                alo[mt] = join8(*(const bf16x8*)&XL[base], *(const bf16x8*)&XL[base + 16]);
            }
            // B frags (32x16 bf16): lanes 0-15 -> K[0..7],[16..23] at N=lane; lanes 16-31 -> K[8..15],[24..31]
            v16bf bhi[4], blo[4];
#pragma unroll
            for (int nt = 0; nt < 4; ++nt) {
                int col  = wn * 64 + nt * 16 + l15;
                int base = col * LDSS + ks + hs * 8;
                bhi[nt] = join8(*(const bf16x8*)&WH[base], *(const bf16x8*)&WH[base + 16]);
                blo[nt] = join8(*(const bf16x8*)&WL[base], *(const bf16x8*)&WL[base + 16]);
            }
#pragma unroll
            for (int mt = 0; mt < 2; ++mt)
#pragma unroll
                for (int nt = 0; nt < 4; ++nt) {
                    v8f c = acc[mt][nt];
                    c = __builtin_amdgcn_wmma_f32_16x16x32_bf16(false, ahi[mt], false, blo[nt],
                                                                (short)0, c, false, false);
                    c = __builtin_amdgcn_wmma_f32_16x16x32_bf16(false, alo[mt], false, bhi[nt],
                                                                (short)0, c, false, false);
                    c = __builtin_amdgcn_wmma_f32_16x16x32_bf16(false, ahi[mt], false, bhi[nt],
                                                                (short)0, c, false, false);
                    acc[mt][nt] = c;
                }
        }
    }

    // ---- store (C/D layout: VGPR r -> M=r (lanes 0-15) / M=r+8 (lanes 16-31), N=l15) ----
#pragma unroll
    for (int mt = 0; mt < 2; ++mt) {
        int mbase = m0 + wm * 32 + mt * 16 + hs * 8;
#pragma unroll
        for (int nt = 0; nt < 4; ++nt) {
            int col = o0 + wn * 64 + nt * 16 + l15;
            v8f c = acc[mt][nt];
#pragma unroll
            for (int r = 0; r < 8; ++r) {
                int m = mbase + r;
                if (m < seg_len)
                    out[(size_t)(seg_beg + m) * ODIM + col] = c[r];
            }
        }
    }
}

extern "C" void kernel_launch(void* const* d_in, const int* in_sizes, int n_in,
                              void* d_out, int out_size, void* d_ws, size_t ws_size,
                              hipStream_t stream) {
    const float* x      = (const float*)d_in[0];
    const float* weight = (const float*)d_in[1];
    const float* bias   = (const float*)d_in[2];
    const int*   net_id = (const int*)d_in[3];
    // d_in[4] (slot_id) unused: offsets recovered by binary search on sorted net_id
    float* out = (float*)d_out;
    int* offs  = (int*)d_ws;   // 513 ints, fully rewritten every call

    seg_offsets_kernel<<<dim3((NNET + 1 + 255) / 256), dim3(256), 0, stream>>>(net_id, offs);

    dim3 grid(NNET, (768 + BM - 1) / BM, ODIM / BN);   // 512 x 6 x 2
    grouped_linear_kernel<<<grid, dim3(256), 0, stream>>>(x, weight, bias, offs, out);
}
